// Img2Col_79980880986325
// MI455X (gfx1250) — matrix operand order repaired and checked by reading, not compiled, not run on hardware
//
#include <hip/hip_runtime.h>
#include <hip/hip_bf16.h>

// ---------------------------------------------------------------------------
// Correlation / cost volume (MAX_DISP=4, D=9) for N=8, C=256, H=W=128.
// Pass 1a: NCHW f32 -> NHWC f16 for fmap0 (unpadded)            [64   MiB]
// Pass 1b: zero halo + NCHW f32 -> padded NHWC f16 for fmap1    [72.3 MiB]
// Pass 2 : per 4x4 output-pixel tile, Gram-tile GEMM
//          D[16x144] = A[16x256] * B[256x144], the 144 region pixels tiled
//          as a 3x3 grid of 4x4 blocks so all 9 B-tile addresses are
//          immediate offsets off one base. Per K-chunk all 9 B fragments
//          are loaded as one batch (deep loadcnt pipeline), then 9 WMMAs.
//          Full Gram tile dumped to LDS with immediate-offset ds_store,
//          then re-gathered coalesced.
// ---------------------------------------------------------------------------

typedef __attribute__((ext_vector_type(16))) _Float16 v16h;
typedef __attribute__((ext_vector_type(8)))  _Float16 v8h;
typedef __attribute__((ext_vector_type(8)))  float    v8f;

#define NN 8
#define CC 256
#define HH 128
#define WW 128
#define DD 9           // 2*MAX_DISP+1
#define RR 12          // region side for a 4x4 tile: 4 + 2*4
#define PH 136         // HH + 8
#define PW 136         // WW + 8

// ---------------- Pass 1a: NCHW f32 -> NHWC f16 (fmap0) ----------------
__global__ __launch_bounds__(256)
void nchw_to_nhwc_f16(const float* __restrict__ src, _Float16* __restrict__ dst) {
    __shared__ float tile[32][33];
    const int nh = blockIdx.z;            // n*H + h
    const int n  = nh >> 7;
    const int h  = nh & 127;
    const int c0 = blockIdx.y * 32;
    const int w0 = blockIdx.x * 32;
    const int tx = threadIdx.x;           // 0..31
    const int ty = threadIdx.y;           // 0..7

#pragma unroll
    for (int i = 0; i < 32; i += 8) {
        const int c = c0 + ty + i;
        tile[ty + i][tx] = src[(((size_t)n * CC + c) * HH + h) * WW + w0 + tx];
    }
    __syncthreads();
#pragma unroll
    for (int i = 0; i < 32; i += 8) {
        const int w = w0 + ty + i;
        dst[(((size_t)nh) * WW + w) * CC + c0 + tx] = (_Float16)tile[tx][ty + i];
    }
}

// ---------------- Pass 1b-1: zero the halo of padded fmap1 ----------------
__global__ __launch_bounds__(256)
void zero_halo_f16(_Float16* __restrict__ dst) {
    const int nph = blockIdx.x;                 // n*PH + ph
    const int ph  = nph % PH;
    const int c   = threadIdx.x;                // 0..255
    const bool rowHalo = (ph < 4) || (ph >= PH - 4);
    for (int pw = 0; pw < PW; ++pw) {
        if (rowHalo || (pw < 4) || (pw >= PW - 4)) {
            dst[(((size_t)nph) * PW + pw) * CC + c] = (_Float16)0.f;
        }
    }
}

// ---------------- Pass 1b-2: NCHW f32 -> padded NHWC f16 (fmap1) ----------------
__global__ __launch_bounds__(256)
void nchw_to_nhwc_pad_f16(const float* __restrict__ src, _Float16* __restrict__ dst) {
    __shared__ float tile[32][33];
    const int nh = blockIdx.z;            // n*H + h
    const int n  = nh >> 7;
    const int h  = nh & 127;
    const int c0 = blockIdx.y * 32;
    const int w0 = blockIdx.x * 32;
    const int tx = threadIdx.x;
    const int ty = threadIdx.y;

#pragma unroll
    for (int i = 0; i < 32; i += 8) {
        const int c = c0 + ty + i;
        tile[ty + i][tx] = src[(((size_t)n * CC + c) * HH + h) * WW + w0 + tx];
    }
    __syncthreads();
#pragma unroll
    for (int i = 0; i < 32; i += 8) {
        const int w = w0 + ty + i;
        dst[(((size_t)(n * PH + h + 4)) * PW + (w + 4)) * CC + c0 + tx] =
            (_Float16)tile[tx][ty + i];
    }
}

// ---------------- Pass 2: WMMA correlation ----------------
// One wave per 4x4 output tile; 8 waves per block.
__global__ __launch_bounds__(256)
void corr_wmma_f16(const _Float16* __restrict__ f0h,   // NHWC f16 [N][H][W][C]
                   const _Float16* __restrict__ f1p,   // padded NHWC f16 [N][PH][PW][C]
                   float* __restrict__ out) {
    // Full Gram tile per wave: 144 region pixels x 16 output pixels.
    __shared__ float sgram[8][RR * RR * 16];           // 8 * 9216 B = 72 KiB

    const int lane = threadIdx.x & 31;
    const int wv   = threadIdx.x >> 5;
    const int tile = blockIdx.x * 8 + wv;              // 0..8191
    const int n    = tile >> 10;
    const int th   = (tile >> 5) & 31;
    const int tw   = tile & 31;
    const int h0   = th * 4;
    const int w0   = tw * 4;

    const int j   = lane & 15;                         // column / A-row selector
    const int hi  = lane >> 4;                         // half-wave selector
    const int jrh = j >> 2;                            // 0..3 (row inside 4x4 block)
    const int jrw = j & 3;                             // 0..3 (col inside 4x4 block)

    // ---- A fragment base: row m = j is output pixel (h0 + m/4, w0 + m%4).
    // Contiguous K order used here (matching B): lanes<16 hold K 0..15, lanes>=16 hold
    // K 16..31  ->  per k-chunk: two v8h loads at +hi*16 and +8+hi*16.
    const _Float16* aRow =
        f0h + ((size_t)(n * HH + h0 + jrh) * WW + (w0 + jrw)) * CC + hi * 16;

    // ---- B base: tile (bh,bw), column j = region pixel (bh*4+jrh, bw*4+jrw)
    // in the 12x12 support window anchored at padded coords (h0, w0).
    // All 9 tile offsets are compile-time constants off this one base.
    // Contiguous K order used here (matching A): lanes<16 hold K k0..k0+15, lanes>=16 hold
    // K k0+16..k0+31 -> one contiguous v16h (32B) load at +hi*16.
    const _Float16* bBase =
        f1p + ((size_t)(n * PH + h0 + jrh) * PW + (w0 + jrw)) * CC + hi * 16;

    v8f acc[9] = {};

#pragma unroll
    for (int k0 = 0; k0 < CC; k0 += 32) {
        // Issue the whole chunk's loads as one independent batch so the
        // WMMAs below retire them under staggered loadcnt waits.
        const v8h alo = *(const v8h*)(aRow + k0);
        const v8h ahi = *(const v8h*)(aRow + k0 + 8);

        v16h b[9];
#pragma unroll
        for (int bh = 0; bh < 3; ++bh) {
#pragma unroll
            for (int bw = 0; bw < 3; ++bw) {
                b[bh * 3 + bw] = *(const v16h*)
                    (bBase + (bh * 4 * PW + bw * 4) * CC + k0);   // imm offset
            }
        }

        const v16h a = __builtin_shufflevector(alo, ahi,
                           0, 1, 2, 3, 4, 5, 6, 7, 8, 9, 10, 11, 12, 13, 14, 15);
#pragma unroll
        for (int nt = 0; nt < 9; ++nt) {
            acc[nt] = __builtin_amdgcn_wmma_f32_16x16x32_f16(
                false, a, false, b[nt], (short)0, acc[nt], false, false);
        }
    }

    // ---- Dump the full 16x144 Gram tile to LDS, slot = r*16 + m.
    // C/D layout: lane holds column N = j, VGPR v holds row M = v + 8*hi.
    // r = (bh*4+jrh)*12 + (bw*4+jrw)  ->  slot index is
    //   [(jrh*12 + jrw)*16 + hi*8]  (per-lane, one address reg)
    // + [(bh*48 + bw*4)*16 + v]     (compile-time immediate).
    float* mys = sgram[wv];
    float* base = &mys[(jrh * RR + jrw) * 16 + hi * 8];
#pragma unroll
    for (int bh = 0; bh < 3; ++bh) {
#pragma unroll
        for (int bw = 0; bw < 3; ++bw) {
#pragma unroll
            for (int v = 0; v < 8; ++v) {
                base[(bh * 4 * RR + bw * 4) * 16 + v] = acc[bh * 3 + bw][v];
            }
        }
    }
    __syncthreads();

    // ---- Gather + coalesced store: [N, 81, H, W].
    // For output (dyx, m): r = (py+dy)*12 + (px+dx).
    const size_t planeHW = (size_t)HH * WW;
    const size_t outBase = (size_t)n * (DD * DD) * planeHW;
    for (int i = lane; i < DD * DD * 16; i += 32) {
        const int dyx = i >> 4;
        const int m   = i & 15;
        const int dy  = dyx / DD;
        const int dx  = dyx - dy * DD;
        const int r   = ((m >> 2) + dy) * RR + (m & 3) + dx;
        const int h   = h0 + (m >> 2);
        const int w   = w0 + (m & 3);
        out[outBase + (size_t)dyx * planeHW + (size_t)h * WW + w] = mys[r * 16 + m];
    }
}

// ---------------------------------------------------------------------------
extern "C" void kernel_launch(void* const* d_in, const int* in_sizes, int n_in,
                              void* d_out, int out_size, void* d_ws, size_t ws_size,
                              hipStream_t stream) {
    const float* f0 = (const float*)d_in[0];
    const float* f1 = (const float*)d_in[1];
    float* out = (float*)d_out;

    const size_t f0Bytes = (size_t)NN * HH * WW * CC * 2;        //  64   MiB
    const size_t f1Bytes = (size_t)NN * PH * PW * CC * 2;        // ~72.3 MiB
    if (ws_size < f0Bytes + f1Bytes) return;                     // need ~136.3 MiB

    _Float16* f0h = (_Float16*)d_ws;
    _Float16* f1p = (_Float16*)((char*)d_ws + f0Bytes);

    // Pass 1: layout transform + f16 convert.
    dim3 tb(32, 8, 1);
    dim3 tg(WW / 32, CC / 32, NN * HH);
    nchw_to_nhwc_f16<<<tg, tb, 0, stream>>>(f0, f0h);
    zero_halo_f16<<<NN * PH, 256, 0, stream>>>(f1p);
    nchw_to_nhwc_pad_f16<<<tg, tb, 0, stream>>>(f1, f1p);

    // Pass 2: 8192 tiles, 1 wave each, 8 waves per block.
    corr_wmma_f16<<<8192 / 8, 256, 0, stream>>>(f0h, f1p, out);
}
